// DPSOM_67542655697079
// MI455X (gfx1250) — matrix operand order repaired and checked by reading, not compiled, not run on hardware
//
#include <hip/hip_runtime.h>

// ---- types ----------------------------------------------------------------
typedef __attribute__((ext_vector_type(16))) __bf16          v16bf;
typedef __attribute__((ext_vector_type(8)))  __bf16          v8bf;
typedef __attribute__((ext_vector_type(8)))  float           v8f;
typedef __attribute__((ext_vector_type(16))) unsigned short  v16u;
typedef __attribute__((ext_vector_type(8)))  unsigned short  v8u;
typedef unsigned short u16;

#define BATCH     65536
#define IN_DIM    784
#define HID       256
#define LAT       32
#define NPROTO    25
#define KPAD      800            // 784 padded to 25*32, zeros in tail
#define ROWS_BLK  32             // rows per workgroup in main kernel

// hardware f32->bf16 (round-to-nearest-even), lowers to v_cvt ops
__device__ __forceinline__ u16 f2bf(float f) {
    return __builtin_bit_cast(u16, (__bf16)f);
}

// ---- prep 1: transpose+convert weights to bf16, prototype norms -----------
__global__ void prep_weights(const float* __restrict__ We1,
                             const float* __restrict__ We2,
                             const float* __restrict__ proto,
                             u16* __restrict__ We1T,        // [256][800]
                             u16* __restrict__ We2T,        // [32][256]
                             float* __restrict__ pnorm) {   // [25]
    int i = blockIdx.x * blockDim.x + threadIdx.x;
    const int N1 = HID * KPAD;             // 204800
    const int N2 = LAT * HID;              // 8192
    if (i < N1) {
        int n = i / KPAD, k = i - n * KPAD;
        We1T[i] = (k < IN_DIM) ? f2bf(We1[(size_t)k * HID + n]) : (u16)0;
    } else if (i < N1 + N2) {
        int j = i - N1;
        int n = j / HID, k = j - n * HID;
        We2T[j] = f2bf(We2[(size_t)k * LAT + n]);
    } else if (i < N1 + N2 + NPROTO) {
        int p = i - (N1 + N2);
        float s = 0.f;
        for (int j = 0; j < LAT; ++j) { float v = proto[p * LAT + j]; s = fmaf(v, v, s); }
        pnorm[p] = s;
    }
}

// ---- prep 2: decoder lookup table (25 x 784) ------------------------------
__global__ void prep_decoder(const float* __restrict__ proto,
                             const float* __restrict__ Wd1, const float* __restrict__ bd1,
                             const float* __restrict__ Wd2, const float* __restrict__ bd2,
                             float* __restrict__ recon) {    // [25][784]
    __shared__ float hd[HID];
    __shared__ float pz[LAT];
    int p = blockIdx.x, t = threadIdx.x;
    if (t < LAT) pz[t] = proto[p * LAT + t];
    __syncthreads();
    float s = bd1[t];
    for (int j = 0; j < LAT; ++j) s = fmaf(pz[j], Wd1[j * HID + t], s);
    hd[t] = fmaxf(s, 0.f);
    __syncthreads();
    for (int o = t; o < IN_DIM; o += 256) {
        float v = bd2[o];
        for (int j = 0; j < HID; ++j) v = fmaf(hd[j], Wd2[(size_t)j * IN_DIM + o], v);
        recon[(size_t)p * IN_DIM + o] = 1.f / (1.f + __expf(-v));
    }
}

// ---- main fused kernel ----------------------------------------------------
// block: 256 threads = 8 waves (wave32); 32 batch rows per block.
__global__ __launch_bounds__(256)
void dpsom_main(const float* __restrict__ x,
                const float* __restrict__ be1,
                const float* __restrict__ be2,
                const float* __restrict__ proto,
                const u16*   __restrict__ We1T,     // [256][800] bf16
                const u16*   __restrict__ We2T,     // [32][256]  bf16
                const float* __restrict__ recon,    // [25][784]
                const float* __restrict__ pnorm,    // [25]
                float* __restrict__ xr_out,         // [B][784]
                float* __restrict__ z_out,          // [B][32]
                float* __restrict__ som_out) {      // [B][32]
    __shared__ __attribute__((aligned(32))) u16   hlds[ROWS_BLK][HID];
    __shared__ __attribute__((aligned(16))) float zlds[ROWS_BLK][LAT];
    __shared__ __attribute__((aligned(16))) float plds[NPROTO * LAT];
    __shared__ float pnl[NPROTO];
    __shared__ int   idxl[ROWS_BLK];

    const int tid    = threadIdx.x;
    const int wave   = tid >> 5;
    const int lane   = tid & 31;
    const int lanelo = lane & 15;
    const int lanehi = lane >> 4;
    const int rowBase = blockIdx.x * ROWS_BLK;

    // prototypes + norms into LDS (consumed after later barriers)
    for (int i = tid; i < NPROTO * LAT; i += 256) plds[i] = proto[i];
    if (tid < NPROTO) pnl[tid] = pnorm[tid];

    // ===== stage 1: h = relu(x @ We1 + be1)  -> hlds (bf16) =====
    // wave w: m-subtile (w&1) [16 rows], n-range (w>>1)*64 .. +63 (4 tiles)
    {
        const int msub  = wave & 1;
        const int nbase = (wave >> 1) * 64;
        const int row   = msub * 16 + lanelo;                 // local row this lane feeds A with
        const float* xrow = x + (size_t)(rowBase + row) * IN_DIM;

        v8f acc[4];
        #pragma unroll
        for (int t = 0; t < 4; ++t) acc[t] = (v8f)0.f;

        for (int ks = 0; ks < 25; ++ks) {
            const int k0 = ks * 32;
            // --- A fragment (ISA 16-bit A 16x32 layout):
            // lanes 0-15: K = k0+{0..7} U {16..23};  lanes 16-31: +8
            const int kA = k0 + (lanehi << 3);
            if (ks < 24) __builtin_prefetch(xrow + kA + 64, 0, 0);
            v8bf lo, hi;
            {
                v8f flo = *(const v8f*)(xrow + kA);           // 32B aligned
                lo = __builtin_convertvector(flo, v8bf);      // v_cvt_pk_bf16_f32 x4
                if (ks < 24) {
                    v8f fhi = *(const v8f*)(xrow + kA + 16);
                    hi = __builtin_convertvector(fhi, v8bf);
                } else {
                    // K = 784..799 region: We1T is zero-padded there, A can be 0
                    hi = (v8bf)(__bf16)0.f;
                }
            }
            v16bf a = __builtin_shufflevector(lo, hi,
                        0,1,2,3,4,5,6,7,8,9,10,11,12,13,14,15);
            // --- B fragments: lane holds column n; lanes 0-15: K=k0+{0..7} U {16..23}, 16-31: +8
            const int kB = k0 + (lanehi << 3);
            #pragma unroll
            for (int t = 0; t < 4; ++t) {
                const u16* bp = We1T + (size_t)(nbase + 16 * t + lanelo) * KPAD + kB;
                v8u b0 = *(const v8u*)bp;
                v8u b1 = *(const v8u*)(bp + 16);
                v16u bu = __builtin_shufflevector(b0, b1, 0,1,2,3,4,5,6,7,8,9,10,11,12,13,14,15);
                v16bf b = __builtin_bit_cast(v16bf, bu);
                acc[t] = __builtin_amdgcn_wmma_f32_16x16x32_bf16(
                    false, a, false, b, (short)0, acc[t], false, false);
            }
        }
        // bias + relu + bf16 -> LDS (C layout: lanes0-15 vgpr r -> M=r; lanes16-31 -> M=8+r)
        #pragma unroll
        for (int t = 0; t < 4; ++t) {
            const int n = nbase + 16 * t + lanelo;
            const float bias = be1[n];
            #pragma unroll
            for (int r = 0; r < 8; ++r) {
                const int m = msub * 16 + lanehi * 8 + r;
                float v = fmaxf(acc[t][r] + bias, 0.f);
                hlds[m][n] = f2bf(v);
            }
        }
    }
    __syncthreads();

    // ===== stage 2: z = h @ We2 + be2 (waves 0-3), write z + zlds =====
    if (wave < 4) {
        const int msub = wave & 1;
        const int nsub = wave >> 1;
        const int row  = msub * 16 + lanelo;
        v8f acc = (v8f)0.f;
        for (int ks = 0; ks < 8; ++ks) {
            const int k0 = ks * 32;
            const int kA = k0 + (lanehi << 4);
            v8u ha0 = *(const v8u*)(&hlds[row][kA]);
            v8u ha1 = *(const v8u*)(&hlds[row][kA + 8]);
            v16u au = __builtin_shufflevector(ha0, ha1,
                        0,1,2,3,4,5,6,7,8,9,10,11,12,13,14,15);
            const int kB = k0 + (lanehi << 4);
            v16u bu = *(const v16u*)(We2T + (size_t)(nsub * 16 + lanelo) * HID + kB);
            acc = __builtin_amdgcn_wmma_f32_16x16x32_bf16(
                false, __builtin_bit_cast(v16bf, au),
                false, __builtin_bit_cast(v16bf, bu),
                (short)0, acc, false, false);
        }
        const int n = nsub * 16 + lanelo;
        const float bias = be2[n];
        #pragma unroll
        for (int r = 0; r < 8; ++r) {
            const int m = msub * 16 + lanehi * 8 + r;
            const float v = acc[r] + bias;
            z_out[(size_t)(rowBase + m) * LAT + n] = v;
            zlds[m][n] = v;
        }
    }
    __syncthreads();

    // ===== stage 3: SOM argmin (wave 0, one row per lane) =====
    if (tid < ROWS_BLK) {
        float zr[LAT];
        #pragma unroll
        for (int j = 0; j < LAT; ++j) zr[j] = zlds[tid][j];
        float best = 3.4e38f; int bi = 0;
        for (int p = 0; p < NPROTO; ++p) {
            float dot = 0.f;
            #pragma unroll
            for (int j = 0; j < LAT; ++j) dot = fmaf(plds[p * LAT + j], zr[j], dot);
            float d = pnl[p] - 2.f * dot;              // ||z||^2 constant: drop
            if (d < best) { best = d; bi = p; }
        }
        idxl[tid] = bi;
    }
    __syncthreads();

    // ===== stage 4: gather som_z and x_recon from tables =====
    {   // som_z: 32 rows x 8 float4
        const int row = tid >> 3;
        const int c0  = (tid & 7) * 4;
        const int p   = idxl[row];
        float4 v = *(const float4*)(&plds[p * LAT + c0]);
        *(float4*)(som_out + (size_t)(rowBase + row) * LAT + c0) = v;
    }
    {   // x_recon: 32 rows x 196 float4 from recon table
        for (int i = tid; i < ROWS_BLK * (IN_DIM / 4); i += 256) {
            const int row = i / (IN_DIM / 4);
            const int c4  = i - row * (IN_DIM / 4);
            const int p   = idxl[row];
            float4 v = *(const float4*)(recon + (size_t)p * IN_DIM + c4 * 4);
            *(float4*)(xr_out + (size_t)(rowBase + row) * IN_DIM + c4 * 4) = v;
        }
    }
}

// ---- host launch ----------------------------------------------------------
extern "C" void kernel_launch(void* const* d_in, const int* in_sizes, int n_in,
                              void* d_out, int out_size, void* d_ws, size_t ws_size,
                              hipStream_t stream) {
    const float* x     = (const float*)d_in[0];
    const float* We1   = (const float*)d_in[1];
    const float* be1   = (const float*)d_in[2];
    const float* We2   = (const float*)d_in[3];
    const float* be2   = (const float*)d_in[4];
    const float* Wd1   = (const float*)d_in[5];
    const float* bd1   = (const float*)d_in[6];
    const float* Wd2   = (const float*)d_in[7];
    const float* bd2   = (const float*)d_in[8];
    const float* proto = (const float*)d_in[9];

    unsigned char* ws = (unsigned char*)d_ws;
    u16*   We1T  = (u16*)(ws);                        // 256*800*2   = 409600 B
    u16*   We2T  = (u16*)(ws + 409600);               // 32*256*2    =  16384 B
    float* recon = (float*)(ws + 425984);             // 25*784*4    =  78400 B
    float* pnorm = (float*)(ws + 504384);             // 25*4

    float* xr  = (float*)d_out;
    float* z   = xr + (size_t)BATCH * IN_DIM;
    float* som = z  + (size_t)BATCH * LAT;

    {
        const int work = HID * KPAD + LAT * HID + NPROTO;
        prep_weights<<<(work + 255) / 256, 256, 0, stream>>>(We1, We2, proto, We1T, We2T, pnorm);
    }
    prep_decoder<<<NPROTO, 256, 0, stream>>>(proto, Wd1, bd1, Wd2, bd2, recon);
    dpsom_main<<<BATCH / ROWS_BLK, 256, 0, stream>>>(x, be1, be2, proto, We1T, We2T,
                                                     recon, pnorm, xr, z, som);
}
